// FluviFormerBlock_30751965839623
// MI455X (gfx1250) — matrix operand order repaired and checked by reading, not compiled, not run on hardware
//
#include <hip/hip_runtime.h>
#include <math.h>

typedef __attribute__((ext_vector_type(16))) __bf16 v16bf;
typedef __attribute__((ext_vector_type(8)))  float  v8f;
typedef __attribute__((ext_vector_type(4)))  unsigned int v4u;
typedef __bf16 bf16_t;

#define TOT   200704      // total tokens = 4*224*224
#define CDIM  96
#define HID   384
#define NQKV  288
#define WINN  49
#define NWIN  4096        // 4 * 32 * 32 windows
#define IMGW  224
#define LPB   50176       // 224*224

union Frag { v16bf bf; v4u u[2]; };

__device__ __forceinline__ float gelu_exact(float v) {
    return 0.5f * v * (1.0f + erff(v * 0.70710678118654752f));
}

// ---------------- weight prep: f32 [N,K] -> bf16 [N,K] ----------------
__global__ void k_transpose_bf16(const float* __restrict__ src, bf16_t* __restrict__ dst,
                                 int N, int K) {
    int idx = blockIdx.x * 256 + threadIdx.x;
    if (idx >= N * K) return;
    int n = idx / K, k = idx - n * K;
    dst[(size_t)n * K + k] = (bf16_t)src[idx];
}

// conv_w (Cout,Cin,3,3) -> bf16 [cout][tap*96+cin]
__global__ void k_convw_bf16(const float* __restrict__ src, bf16_t* __restrict__ dst) {
    int idx = blockIdx.x * 256 + threadIdx.x;
    if (idx >= 96 * 96 * 9) return;
    int cout = idx / 864;
    int r = idx - cout * 864;
    int cin = r / 9, tap = r - cin * 9;
    dst[(size_t)cout * 864 + tap * 96 + cin] = (bf16_t)src[idx];
}

// ---------------- LN1 + roll(+3,+3) + window partition -> bf16, + mask vals ----
__global__ __launch_bounds__(256) void k_ln1_window(
    const float* __restrict__ x, const float* __restrict__ ndwi,
    const float* __restrict__ g, const float* __restrict__ b,
    bf16_t* __restrict__ xw, float* __restrict__ mwin) {
    int wave = threadIdx.x >> 5, lane = threadIdx.x & 31;
    long gi = (long)blockIdx.x * 8 + wave;
    if (gi >= TOT) return;
    int w = (int)(gi / WINN), n = (int)(gi - (long)w * WINN);
    int bb = w >> 10, wy = (w >> 5) & 31, wx = w & 31;
    int ty = n / 7, tx = n - ty * 7;
    int ys = wy * 7 + ty, xs = wx * 7 + tx;
    int y = ys - 3; if (y < 0) y += IMGW;
    int xp = xs - 3; if (xp < 0) xp += IMGW;
    size_t t = (size_t)bb * LPB + (size_t)y * IMGW + xp;
    const float* row = x + t * CDIM;
    float v0 = row[lane], v1 = row[lane + 32], v2 = row[lane + 64];
    float s = v0 + v1 + v2;
    for (int o = 16; o; o >>= 1) s += __shfl_xor(s, o, 32);
    float mean = s * (1.0f / 96.0f);
    float d0 = v0 - mean, d1 = v1 - mean, d2 = v2 - mean;
    float vs = d0 * d0 + d1 * d1 + d2 * d2;
    for (int o = 16; o; o >>= 1) vs += __shfl_xor(vs, o, 32);
    float inv = rsqrtf(vs * (1.0f / 96.0f) + 1e-5f);
    bf16_t* orow = xw + (size_t)gi * CDIM;
    orow[lane]      = (bf16_t)(d0 * inv * g[lane]      + b[lane]);
    orow[lane + 32] = (bf16_t)(d1 * inv * g[lane + 32] + b[lane + 32]);
    orow[lane + 64] = (bf16_t)(d2 * inv * g[lane + 64] + b[lane + 64]);
    if (lane == 0) mwin[gi] = ndwi[t];
}

// ---------------- plain LN (token-order rows) -> bf16 ----------------
__global__ __launch_bounds__(256) void k_ln_plain(
    const float* __restrict__ x, const float* __restrict__ g,
    const float* __restrict__ b, bf16_t* __restrict__ outb) {
    int wave = threadIdx.x >> 5, lane = threadIdx.x & 31;
    long t = (long)blockIdx.x * 8 + wave;
    if (t >= TOT) return;
    const float* row = x + (size_t)t * CDIM;
    float v0 = row[lane], v1 = row[lane + 32], v2 = row[lane + 64];
    float s = v0 + v1 + v2;
    for (int o = 16; o; o >>= 1) s += __shfl_xor(s, o, 32);
    float mean = s * (1.0f / 96.0f);
    float d0 = v0 - mean, d1 = v1 - mean, d2 = v2 - mean;
    float vs = d0 * d0 + d1 * d1 + d2 * d2;
    for (int o = 16; o; o >>= 1) vs += __shfl_xor(vs, o, 32);
    float inv = rsqrtf(vs * (1.0f / 96.0f) + 1e-5f);
    bf16_t* orow = outb + (size_t)t * CDIM;
    orow[lane]      = (bf16_t)(d0 * inv * g[lane]      + b[lane]);
    orow[lane + 32] = (bf16_t)(d1 * inv * g[lane + 32] + b[lane + 32]);
    orow[lane + 64] = (bf16_t)(d2 * inv * g[lane + 64] + b[lane + 64]);
}

// ------- generic WMMA bf16 GEMM: C = A[M,K] * Wt[N,K]^T + bias -----------------
// Wave tile 16x48 (one A fragment feeds 3 WMMAs); block = 8 waves -> 64x96.
// MODE 0: QKV     -> outb bf16, row-major [M,N]
// MODE 1: proj    -> window-reverse + roll(-4) scatter; outf[t] = resid[t] + v
// MODE 2: mlp1    -> outb bf16 = GELU(v)
// MODE 3: mlp2    -> outf[row] = resid[row] + v
template <int MODE>
__global__ __launch_bounds__(256) void k_gemm_wmma(
    const bf16_t* __restrict__ A, const bf16_t* __restrict__ Wt,
    const float* __restrict__ bias, const float* __restrict__ resid,
    float* __restrict__ outf, bf16_t* __restrict__ outb, int K, int N) {
    int wave = threadIdx.x >> 5, lane = threadIdx.x & 31;
    int wm = wave & 3, wn = wave >> 2;
    int m0 = blockIdx.x * 64 + wm * 16;
    int n0 = blockIdx.y * 96 + wn * 48;

    v8f acc0 = {0.f, 0.f, 0.f, 0.f, 0.f, 0.f, 0.f, 0.f};
    v8f acc1 = acc0, acc2 = acc0;
    int mrow = m0 + (lane & 15);
    int kh8 = (lane >> 4) << 3;                 // 0 or 8
    const bf16_t* arow = A + (size_t)mrow * K + kh8;

    for (int kb = 0; kb < K; kb += 32) {
        Frag af, b0, b1, b2;
        af.u[0] = *(const v4u*)(const void*)(arow + kb);
        af.u[1] = *(const v4u*)(const void*)(arow + kb + 16);
        const bf16_t* bp = Wt + (size_t)(n0 + (lane & 15)) * K + kb + kh8;
        b0.u[0] = *(const v4u*)(const void*)(bp);
        b0.u[1] = *(const v4u*)(const void*)(bp + 16);
        b1.u[0] = *(const v4u*)(const void*)(bp + (size_t)16 * K);
        b1.u[1] = *(const v4u*)(const void*)(bp + (size_t)16 * K + 16);
        b2.u[0] = *(const v4u*)(const void*)(bp + (size_t)32 * K);
        b2.u[1] = *(const v4u*)(const void*)(bp + (size_t)32 * K + 16);
        acc0 = __builtin_amdgcn_wmma_f32_16x16x32_bf16(
            false, af.bf, false, b0.bf, (short)0, acc0, false, false);
        acc1 = __builtin_amdgcn_wmma_f32_16x16x32_bf16(
            false, af.bf, false, b1.bf, (short)0, acc1, false, false);
        acc2 = __builtin_amdgcn_wmma_f32_16x16x32_bf16(
            false, af.bf, false, b2.bf, (short)0, acc2, false, false);
    }

    int nn = lane & 15, mh = lane >> 4;
#pragma unroll
    for (int r = 0; r < 8; ++r) {
        int row = m0 + mh * 8 + r;
        size_t t = 0;
        if (MODE == 1) {
            int w = row / WINN, n = row - w * WINN;
            int bb = w >> 10, wy = (w >> 5) & 31, wx = w & 31;
            int ty = n / 7, tx = n - ty * 7;
            int ys = wy * 7 + ty, xs = wx * 7 + tx;
            int y = ys - 4; if (y < 0) y += IMGW;
            int xp = xs - 4; if (xp < 0) xp += IMGW;
            t = (size_t)bb * LPB + (size_t)y * IMGW + xp;
        }
#pragma unroll
        for (int c = 0; c < 3; ++c) {
            int col = n0 + 16 * c + nn;
            float v = (c == 0 ? acc0[r] : (c == 1 ? acc1[r] : acc2[r])) + bias[col];
            if (MODE == 0) {
                outb[(size_t)row * N + col] = (bf16_t)v;
            } else if (MODE == 1) {
                outf[t * CDIM + col] = resid[t * CDIM + col] + v;
            } else if (MODE == 2) {
                outb[(size_t)row * N + col] = (bf16_t)gelu_exact(v);
            } else {                               // MODE 3
                outf[(size_t)row * N + col] = resid[(size_t)row * N + col] + v;
            }
        }
    }
}

// ---------------- windowed attention: one block per (window, head) -----------
__global__ __launch_bounds__(64) void k_attn(
    const bf16_t* __restrict__ qkv, const float* __restrict__ rpb,
    const float* __restrict__ mwin, bf16_t* __restrict__ outb) {
    int blk = blockIdx.x;
    int w = blk >> 3, h = blk & 7;
    __shared__ float qs[WINN * 12], ks[WINN * 12], vs[WINN * 12], ms[WINN];
    int tid = threadIdx.x;
    for (int idx = tid; idx < WINN * 12; idx += 64) {
        int n = idx / 12, d = idx - n * 12;
        size_t base = ((size_t)w * WINN + n) * NQKV + h * 12 + d;
        qs[idx] = (float)qkv[base];
        ks[idx] = (float)qkv[base + 96];
        vs[idx] = (float)qkv[base + 192];
    }
    for (int idx = tid; idx < WINN; idx += 64) ms[idx] = mwin[(size_t)w * WINN + idx];
    __syncthreads();
    if (tid >= WINN) return;

    const float scale = 0.2886751345948129f;  // 12^-0.5
    float q[12];
#pragma unroll
    for (int d = 0; d < 12; ++d) q[d] = qs[tid * 12 + d] * scale;
    int yi = tid / 7, xi = tid - yi * 7;
    float mi = ms[tid];

    float mx = -1e30f;
    for (int j = 0; j < WINN; ++j) {
        float s = 0.f;
#pragma unroll
        for (int d = 0; d < 12; ++d) s += q[d] * ks[j * 12 + d];
        int yj = j / 7, xj = j - yj * 7;
        int ridx = (yi - yj + 6) * 13 + (xi - xj + 6);
        s += rpb[ridx * 8 + h];
        if (ms[j] != mi) s -= 100.0f;
        mx = fmaxf(mx, s);
    }
    float den = 0.f;
    float acc[12];
#pragma unroll
    for (int d = 0; d < 12; ++d) acc[d] = 0.f;
    for (int j = 0; j < WINN; ++j) {
        float s = 0.f;
#pragma unroll
        for (int d = 0; d < 12; ++d) s += q[d] * ks[j * 12 + d];
        int yj = j / 7, xj = j - yj * 7;
        int ridx = (yi - yj + 6) * 13 + (xi - xj + 6);
        s += rpb[ridx * 8 + h];
        if (ms[j] != mi) s -= 100.0f;
        float p = __expf(s - mx);
        den += p;
#pragma unroll
        for (int d = 0; d < 12; ++d) acc[d] += p * vs[j * 12 + d];
    }
    float inv = 1.0f / den;
    size_t ob = ((size_t)w * WINN + tid) * CDIM + h * 12;
#pragma unroll
    for (int d = 0; d < 12; ++d) outb[ob + d] = (bf16_t)(acc[d] * inv);
}

// ---------------- implicit-GEMM 3x3 conv via WMMA (NHWC bf16, K = 864) -------
// Wave tile 16x48; N = 96 -> single grid.y, one pass over the image.
// FINAL 0: out = bf16 GELU(BN(conv + cbias))
// FINAL 1: outf[t] = resid[t] + GELU(BN(conv + cbias))   (resid == outf == d_out)
template <int FINAL>
__global__ __launch_bounds__(256) void k_conv_wmma(
    const bf16_t* __restrict__ img, const bf16_t* __restrict__ Wt,
    const float* __restrict__ cbias,
    const float* __restrict__ bg, const float* __restrict__ bb_,
    const float* __restrict__ bm, const float* __restrict__ bv,
    const float* __restrict__ resid, float* __restrict__ outf,
    bf16_t* __restrict__ outb, int dil) {
    int wave = threadIdx.x >> 5, lane = threadIdx.x & 31;
    int wm = wave & 3, wn = wave >> 2;
    int m0 = blockIdx.x * 64 + wm * 16;
    int n0 = wn * 48;

    v8f acc0 = {0.f, 0.f, 0.f, 0.f, 0.f, 0.f, 0.f, 0.f};
    v8f acc1 = acc0, acc2 = acc0;
    int mrow = m0 + (lane & 15);
    int bidx = mrow / LPB;
    int rem = mrow - bidx * LPB;
    int y = rem / IMGW, x = rem - y * IMGW;
    int kh8 = (lane >> 4) << 3;

    for (int kb = 0; kb < 864; kb += 32) {
        int tap = kb / 96;
        int c0 = kb - tap * 96;
        int dy = (tap / 3 - 1) * dil, dx = (tap % 3 - 1) * dil;
        int yy = y + dy, xx = x + dx;
        Frag af, b0, b1, b2;
        af.u[0] = (v4u){0u, 0u, 0u, 0u};
        af.u[1] = (v4u){0u, 0u, 0u, 0u};
        if ((unsigned)yy < (unsigned)IMGW && (unsigned)xx < (unsigned)IMGW) {
            const bf16_t* p = img + ((size_t)bidx * LPB + (size_t)yy * IMGW + xx) * CDIM
                              + c0 + kh8;
            af.u[0] = *(const v4u*)(const void*)(p);
            af.u[1] = *(const v4u*)(const void*)(p + 16);
        }
        const bf16_t* bp = Wt + (size_t)(n0 + (lane & 15)) * 864 + kb + kh8;
        b0.u[0] = *(const v4u*)(const void*)(bp);
        b0.u[1] = *(const v4u*)(const void*)(bp + 16);
        b1.u[0] = *(const v4u*)(const void*)(bp + 16 * 864);
        b1.u[1] = *(const v4u*)(const void*)(bp + 16 * 864 + 16);
        b2.u[0] = *(const v4u*)(const void*)(bp + 32 * 864);
        b2.u[1] = *(const v4u*)(const void*)(bp + 32 * 864 + 16);
        acc0 = __builtin_amdgcn_wmma_f32_16x16x32_bf16(
            false, af.bf, false, b0.bf, (short)0, acc0, false, false);
        acc1 = __builtin_amdgcn_wmma_f32_16x16x32_bf16(
            false, af.bf, false, b1.bf, (short)0, acc1, false, false);
        acc2 = __builtin_amdgcn_wmma_f32_16x16x32_bf16(
            false, af.bf, false, b2.bf, (short)0, acc2, false, false);
    }

    int nn = lane & 15, mh = lane >> 4;
#pragma unroll
    for (int r = 0; r < 8; ++r) {
        int row = m0 + mh * 8 + r;
#pragma unroll
        for (int c = 0; c < 3; ++c) {
            int col = n0 + 16 * c + nn;
            float s = bg[col] * rsqrtf(bv[col] + 1e-5f);
            float v = (c == 0 ? acc0[r] : (c == 1 ? acc1[r] : acc2[r])) + cbias[col];
            float bn = (v - bm[col]) * s + bb_[col];
            float gv = gelu_exact(bn);
            if (FINAL) outf[(size_t)row * CDIM + col] = resid[(size_t)row * CDIM + col] + gv;
            else       outb[(size_t)row * CDIM + col] = (bf16_t)gv;
        }
    }
}

// ---------------------------------------------------------------------------
extern "C" void kernel_launch(void* const* d_in, const int* in_sizes, int n_in,
                              void* d_out, int out_size, void* d_ws, size_t ws_size,
                              hipStream_t stream) {
    const float* x       = (const float*)d_in[0];
    const float* ndwi    = (const float*)d_in[1];
    const float* n1g     = (const float*)d_in[2];
    const float* n1b     = (const float*)d_in[3];
    const float* qkv_w   = (const float*)d_in[4];
    const float* qkv_bi  = (const float*)d_in[5];
    const float* proj_w  = (const float*)d_in[6];
    const float* proj_b  = (const float*)d_in[7];
    const float* rpb     = (const float*)d_in[8];
    const float* n2g     = (const float*)d_in[9];
    const float* n2b     = (const float*)d_in[10];
    const float* w1      = (const float*)d_in[11];
    const float* b1      = (const float*)d_in[12];
    const float* w2      = (const float*)d_in[13];
    const float* b2      = (const float*)d_in[14];
    const float* n3g     = (const float*)d_in[15];
    const float* n3b     = (const float*)d_in[16];
    const float* c1w     = (const float*)d_in[17];
    const float* c1b     = (const float*)d_in[18];
    const float* bn1g    = (const float*)d_in[19];
    const float* bn1b    = (const float*)d_in[20];
    const float* bn1m    = (const float*)d_in[21];
    const float* bn1v    = (const float*)d_in[22];
    const float* c2w     = (const float*)d_in[23];
    const float* c2b     = (const float*)d_in[24];
    const float* bn2g    = (const float*)d_in[25];
    const float* bn2b    = (const float*)d_in[26];
    const float* bn2m    = (const float*)d_in[27];
    const float* bn2v    = (const float*)d_in[28];
    float* out = (float*)d_out;

    char* ws = (char*)d_ws;
    size_t off = 0;
    auto alloc = [&](size_t bytes) {
        size_t r = off;
        off += (bytes + 255) & ~(size_t)255;
        return r;
    };
    // big region: qkv bf16 (TOT*288*2) reused as mlp-hidden bf16 (TOT*384*2)
    size_t oBig  = alloc((size_t)TOT * HID * 2);
    size_t oX1   = alloc((size_t)TOT * CDIM * 4);
    size_t oBufA = alloc((size_t)TOT * CDIM * 2);   // xw / h / img
    size_t oBufB = alloc((size_t)TOT * CDIM * 2);   // attnout / conv1-out
    size_t oMask = alloc((size_t)TOT * 4);
    size_t oWq   = alloc((size_t)96 * 288 * 2);
    size_t oWp   = alloc((size_t)96 * 96 * 2);
    size_t oW1   = alloc((size_t)96 * 384 * 2);
    size_t oW2   = alloc((size_t)384 * 96 * 2);
    size_t oC1   = alloc((size_t)864 * 96 * 2);
    size_t oC2   = alloc((size_t)864 * 96 * 2);
    (void)ws_size; (void)in_sizes; (void)n_in; (void)out_size;

    bf16_t* qkvB  = (bf16_t*)(ws + oBig);
    bf16_t* hidB  = (bf16_t*)(ws + oBig);
    float*  x1    = (float*)(ws + oX1);
    bf16_t* bufA  = (bf16_t*)(ws + oBufA);
    bf16_t* bufB  = (bf16_t*)(ws + oBufB);
    float*  mwin  = (float*)(ws + oMask);
    bf16_t* WqT   = (bf16_t*)(ws + oWq);
    bf16_t* WpT   = (bf16_t*)(ws + oWp);
    bf16_t* W1T   = (bf16_t*)(ws + oW1);
    bf16_t* W2T   = (bf16_t*)(ws + oW2);
    bf16_t* C1T   = (bf16_t*)(ws + oC1);
    bf16_t* C2T   = (bf16_t*)(ws + oC2);

    // --- weight prep ---
    k_transpose_bf16<<<(288 * 96 + 255) / 256, 256, 0, stream>>>(qkv_w, WqT, 288, 96);
    k_transpose_bf16<<<(96 * 96 + 255) / 256, 256, 0, stream>>>(proj_w, WpT, 96, 96);
    k_transpose_bf16<<<(384 * 96 + 255) / 256, 256, 0, stream>>>(w1, W1T, 384, 96);
    k_transpose_bf16<<<(96 * 384 + 255) / 256, 256, 0, stream>>>(w2, W2T, 96, 384);
    k_convw_bf16<<<(96 * 96 * 9 + 255) / 256, 256, 0, stream>>>(c1w, C1T);
    k_convw_bf16<<<(96 * 96 * 9 + 255) / 256, 256, 0, stream>>>(c2w, C2T);

    const int MT = TOT / 64;   // 3136 M-tile blocks; grid.y = N / 96

    // --- attention path ---
    k_ln1_window<<<TOT / 8, 256, 0, stream>>>(x, ndwi, n1g, n1b, bufA, mwin);
    k_gemm_wmma<0><<<dim3(MT, NQKV / 96), 256, 0, stream>>>(
        bufA, WqT, qkv_bi, nullptr, nullptr, qkvB, 96, NQKV);
    k_attn<<<NWIN * 8, 64, 0, stream>>>(qkvB, rpb, mwin, bufB);
    k_gemm_wmma<1><<<dim3(MT, 1), 256, 0, stream>>>(
        bufB, WpT, proj_b, x, x1, nullptr, 96, 96);

    // --- MLP path ---
    k_ln_plain<<<TOT / 8, 256, 0, stream>>>(x1, n2g, n2b, bufA);
    k_gemm_wmma<2><<<dim3(MT, HID / 96), 256, 0, stream>>>(
        bufA, W1T, b1, nullptr, nullptr, hidB, 96, HID);
    k_gemm_wmma<3><<<dim3(MT, 1), 256, 0, stream>>>(
        hidB, W2T, b2, x1, out, nullptr, HID, 96);

    // --- multi-scale CNN branch ---
    k_ln_plain<<<TOT / 8, 256, 0, stream>>>(out, n3g, n3b, bufA);
    k_conv_wmma<0><<<dim3(MT, 1), 256, 0, stream>>>(
        bufA, C1T, c1b, bn1g, bn1b, bn1m, bn1v, nullptr, nullptr, bufB, 1);
    k_conv_wmma<1><<<dim3(MT, 1), 256, 0, stream>>>(
        bufB, C2T, c2b, bn2g, bn2b, bn2m, bn2v, out, out, nullptr, 2);
}
